// Block_47321949667759
// MI455X (gfx1250) — compile-verified, matrix operand order checked by reading, not run on hardware
//
#include <hip/hip_runtime.h>
#include <hip/hip_bf16.h>

typedef __attribute__((ext_vector_type(16))) __bf16 v16bf;
typedef __attribute__((ext_vector_type(8)))  float  v8f;

union BFrag {            // 8 VGPRs = 16 bf16 elements
    v16bf v;
    uint4 u[2];
};

#define Ecfg 1024
#define Hcfg 16
#define Dcfg 64
#define Tcfg 2048
#define Bcfg 2
#define Mcfg (Bcfg * Tcfg)   // 4096 tokens

// ---- CDNA5 async global->LDS copy (ASYNCcnt-tracked, VGPR-bypassing) ------
#if defined(__has_builtin)
#  if __has_builtin(__builtin_amdgcn_global_load_async_to_lds_b128)
#    define HAVE_ASYNC_COPY 1
#  endif
#  if __has_builtin(__builtin_amdgcn_s_wait_asynccnt)
#    define HAVE_WAIT_ASYNC 1
#  endif
#endif
#ifndef HAVE_ASYNC_COPY
#  define HAVE_ASYNC_COPY 0
#endif
#ifndef HAVE_WAIT_ASYNC
#  define HAVE_WAIT_ASYNC 0
#endif

template <int N>
__device__ __forceinline__ void wait_asynccnt() {
#if HAVE_WAIT_ASYNC
    __builtin_amdgcn_s_wait_asynccnt(N);
#else
    asm volatile("s_wait_asynccnt %0" ::"n"(N) : "memory");
#endif
}

#if HAVE_ASYNC_COPY
typedef int v4i_ __attribute__((vector_size(16)));   // matches builtin pointee type
__device__ __forceinline__ void async_copy16(const void* g, void* l) {
    __builtin_amdgcn_global_load_async_to_lds_b128(
        (__attribute__((address_space(1))) v4i_*)g,
        (__attribute__((address_space(3))) v4i_*)l, 0, 0);
}
#endif

__device__ __forceinline__ float gelu_tanh(float x) {
    float x3 = x * x * x;
    return 0.5f * x * (1.0f + tanhf(0.79788456080286535588f * (x + 0.044715f * x3)));
}

// ---------------------------------------------------------------------------
// Weight convert: w[K,N] f32  ->  wT[N,K] bf16
// ---------------------------------------------------------------------------
__global__ void cvt_wT_kernel(const float* __restrict__ w, __bf16* __restrict__ wT,
                              int K, int N) {
    size_t id = (size_t)blockIdx.x * 256 + threadIdx.x;
    if (id >= (size_t)K * N) return;
    int k = (int)(id / N);
    int n = (int)(id % N);
    wT[(size_t)n * K + k] = (__bf16)w[id];
}

// ---------------------------------------------------------------------------
// LayerNorm: x[M,E] f32 -> out[M,E] bf16   (one block per row)
// ---------------------------------------------------------------------------
__global__ __launch_bounds__(256) void ln_kernel(const float* __restrict__ x,
                                                 const float* __restrict__ g,
                                                 const float* __restrict__ b,
                                                 __bf16* __restrict__ out) {
    int row = blockIdx.x;
    const float* xr = x + (size_t)row * Ecfg;
    float s = 0.f, ss = 0.f;
    for (int i = threadIdx.x; i < Ecfg; i += 256) {
        float v = xr[i];
        s += v;
        ss += v * v;
    }
#pragma unroll
    for (int m = 16; m >= 1; m >>= 1) {
        s  += __shfl_xor(s,  m, 32);
        ss += __shfl_xor(ss, m, 32);
    }
    __shared__ float sh[16];
    int wv = threadIdx.x >> 5, lane = threadIdx.x & 31;
    if (lane == 0) { sh[wv] = s; sh[8 + wv] = ss; }
    __syncthreads();
    if (threadIdx.x == 0) {
        float a = 0.f, c = 0.f;
        for (int i = 0; i < 8; ++i) { a += sh[i]; c += sh[8 + i]; }
        sh[0] = a; sh[8] = c;
    }
    __syncthreads();
    float mu  = sh[0] * (1.0f / Ecfg);
    float var = sh[8] * (1.0f / Ecfg) - mu * mu;
    float inv = rsqrtf(var + 1e-5f);
    for (int i = threadIdx.x; i < Ecfg; i += 256)
        out[(size_t)row * Ecfg + i] = (__bf16)((xr[i] - mu) * inv * g[i] + b[i]);
}

// ---------------------------------------------------------------------------
// WMMA GEMM:  Out[M,N] = A[M,K](bf16) * Wt[N,K]^T(bf16) + bias (+res / gelu)
// Block tile 128x128, 8 waves (2x4), wave tile 64x32, K step 32.
// Double-buffered LDS with async global->LDS copies (ASYNCcnt pipelined).
// EPI: 0 = bf16 out, 1 = f32 out + residual, 2 = gelu -> bf16 out
// ---------------------------------------------------------------------------
template <int EPI>
__global__ __launch_bounds__(256) void gemm_kernel(const __bf16* __restrict__ A,
                                                   const __bf16* __restrict__ Bt,
                                                   const float* __restrict__ bias,
                                                   const float* __restrict__ res,
                                                   void* __restrict__ Out,
                                                   int M, int N, int K) {
#if HAVE_ASYNC_COPY
    __shared__ __align__(16) unsigned short As[2][128 * 40];
    __shared__ __align__(16) unsigned short Bs[2][128 * 40];
#else
    __shared__ __align__(16) unsigned short As[1][128 * 40];
    __shared__ __align__(16) unsigned short Bs[1][128 * 40];
#endif

    int tid = threadIdx.x;
    int lane = tid & 31, wv = tid >> 5;
    int wm = wv >> 2, wn = wv & 3;       // 2 x 4 wave grid
    int lr = lane & 15, lh = lane >> 4;  // lane row / K-half
    int m0 = blockIdx.y * 128;
    int n0 = blockIdx.x * 128;

    v8f vz = {0.f, 0.f, 0.f, 0.f, 0.f, 0.f, 0.f, 0.f};
    v8f acc[4][2];
#pragma unroll
    for (int mi = 0; mi < 4; ++mi)
#pragma unroll
        for (int ni = 0; ni < 2; ++ni) acc[mi][ni] = vz;

    const int ntiles = K >> 5;  // K / 32

#if HAVE_ASYNC_COPY
    // Issue a full 128x32 A tile + 128x32 B tile: 4 async b128 copies/thread.
    auto issue_tile = [&](int k0, int buf) {
#pragma unroll
        for (int i = 0; i < 2; ++i) {
            int idx = tid + 256 * i;
            int row = idx >> 2, ch = idx & 3;
            async_copy16(A  + (size_t)(m0 + row) * K + k0 + ch * 8,
                         &As[buf][row * 40 + ch * 8]);
            async_copy16(Bt + (size_t)(n0 + row) * K + k0 + ch * 8,
                         &Bs[buf][row * 40 + ch * 8]);
        }
    };
    issue_tile(0, 0);
    for (int t = 0; t < ntiles; ++t) {
        int buf = t & 1;
        if (t + 1 < ntiles) {
            issue_tile((t + 1) << 5, (t + 1) & 1);  // prefetch next tile
            wait_asynccnt<4>();                     // oldest 4 (tile t) done
        } else {
            wait_asynccnt<0>();
        }
        __syncthreads();
#else
    for (int t = 0; t < ntiles; ++t) {
        int buf = 0;
        int k0 = t << 5;
#pragma unroll
        for (int i = 0; i < 2; ++i) {
            int idx = tid + 256 * i;
            int row = idx >> 2, ch = idx & 3;
            uint4 va = *(const uint4*)(A  + (size_t)(m0 + row) * K + k0 + ch * 8);
            *(uint4*)&As[0][row * 40 + ch * 8] = va;
            uint4 vb = *(const uint4*)(Bt + (size_t)(n0 + row) * K + k0 + ch * 8);
            *(uint4*)&Bs[0][row * 40 + ch * 8] = vb;
        }
        __syncthreads();
#endif
        BFrag af[4], bf[2];
#pragma unroll
        for (int mi = 0; mi < 4; ++mi) {
            int r = wm * 64 + mi * 16 + lr;
            af[mi].u[0] = *(const uint4*)&As[buf][r * 40 + lh * 8];
            af[mi].u[1] = *(const uint4*)&As[buf][r * 40 + 16 + lh * 8];
        }
#pragma unroll
        for (int ni = 0; ni < 2; ++ni) {
            int r = wn * 32 + ni * 16 + lr;
            bf[ni].u[0] = *(const uint4*)&Bs[buf][r * 40 + lh * 8];
            bf[ni].u[1] = *(const uint4*)&Bs[buf][r * 40 + 16 + lh * 8];
        }
#pragma unroll
        for (int mi = 0; mi < 4; ++mi)
#pragma unroll
            for (int ni = 0; ni < 2; ++ni)
                acc[mi][ni] = __builtin_amdgcn_wmma_f32_16x16x32_bf16(
                    false, af[mi].v, false, bf[ni].v, (short)0, acc[mi][ni], false, false);
        __syncthreads();
    }

    // Epilogue
#pragma unroll
    for (int mi = 0; mi < 4; ++mi) {
#pragma unroll
        for (int ni = 0; ni < 2; ++ni) {
            int n = n0 + wn * 32 + ni * 16 + lr;
            float bn = bias[n];
            int mb = m0 + wm * 64 + mi * 16 + lh * 8;
#pragma unroll
            for (int r = 0; r < 8; ++r) {
                size_t off = (size_t)(mb + r) * N + n;
                float v = acc[mi][ni][r] + bn;
                if constexpr (EPI == 0) {
                    ((__bf16*)Out)[off] = (__bf16)v;
                } else if constexpr (EPI == 1) {
                    ((float*)Out)[off] = v + res[off];
                } else {
                    ((__bf16*)Out)[off] = (__bf16)gelu_tanh(v);
                }
            }
        }
    }
}

// ---------------------------------------------------------------------------
// Split fused qkv[M,3E] (head-interleaved q,k,v per reference reshape) into
// Q[B,H,T,D], K[B,H,T,D], Vt[B,H,D,T]  (V transposed for PV B^T loads)
// ---------------------------------------------------------------------------
__global__ void split_qkv_kernel(const __bf16* __restrict__ qkv,
                                 __bf16* __restrict__ Q, __bf16* __restrict__ Kk,
                                 __bf16* __restrict__ Vt) {
    size_t id = (size_t)blockIdx.x * 256 + threadIdx.x;  // over M*E
    int m = (int)(id >> 10);
    int e = (int)(id & 1023);
    int h = e >> 6, d = e & 63;
    int b = m / Tcfg, t = m % Tcfg;
    int bh = b * Hcfg + h;
    const __bf16* src = qkv + (size_t)m * (3 * Ecfg) + h * (3 * Dcfg) + d;
    Q [((size_t)bh * Tcfg + t) * Dcfg + d] = src[0];
    Kk[((size_t)bh * Tcfg + t) * Dcfg + d] = src[Dcfg];
    Vt[((size_t)bh * Dcfg + d) * Tcfg + t] = src[2 * Dcfg];
}

// ---------------------------------------------------------------------------
// Flash attention (causal). One wave per 16-query tile of one (b,h).
// Computes S^T = K_tile @ Q^T so softmax is per-lane (column = query),
// then O^T = V^T @ P^T; the exp'd S^T C-fragments pack in-lane into the
// B-fragment layout for the PV WMMA (no cross-lane shuffles).
// ---------------------------------------------------------------------------
__global__ __launch_bounds__(256) void attn_kernel(const __bf16* __restrict__ Q,
                                                   const __bf16* __restrict__ Kk,
                                                   const __bf16* __restrict__ Vt,
                                                   __bf16* __restrict__ Outp) {
    int lane = threadIdx.x & 31;
    int wid = (int)((blockIdx.x * 256 + threadIdx.x) >> 5);
    const int qtiles = Tcfg >> 4;  // 128
    int bh = wid / qtiles;
    int qt = wid % qtiles;
    int q_base = qt * 16;
    int lr = lane & 15, lh = lane >> 4;
    int q = q_base + lr;  // query column owned by this lane

    // Q^T B-fragments for the two 32-wide d chunks
    const __bf16* Qp = Q + ((size_t)bh * Tcfg + q) * Dcfg;
    BFrag bq[2];
#pragma unroll
    for (int c = 0; c < 2; ++c) {
        bq[c].u[0] = *(const uint4*)(Qp + c * 32 + lh * 8);
        bq[c].u[1] = *(const uint4*)(Qp + c * 32 + 16 + lh * 8);
    }

    v8f vz = {0.f, 0.f, 0.f, 0.f, 0.f, 0.f, 0.f, 0.f};
    v8f acc[4];  // O^T: 4 d-tiles x (16d x 16q)
#pragma unroll
    for (int dt = 0; dt < 4; ++dt) acc[dt] = vz;
    float mrun = -1e30f, srun = 0.f;

    int kcmax = (q_base + 15) >> 5;
    for (int kc = 0; kc <= kcmax; ++kc) {
        int ktb = kc << 5;
        // S^T tiles: [16kt x 16q] for kt-sub 0/1, K-dim = D = 64 (2 wmmas each)
        v8f st[2];
#pragma unroll
        for (int sub = 0; sub < 2; ++sub) {
            const __bf16* Kp = Kk + ((size_t)bh * Tcfg + ktb + sub * 16 + lr) * Dcfg;
            BFrag ka0, ka1;
            ka0.u[0] = *(const uint4*)(Kp + lh * 8);
            ka0.u[1] = *(const uint4*)(Kp + 16 + lh * 8);
            ka1.u[0] = *(const uint4*)(Kp + 32 + lh * 8);
            ka1.u[1] = *(const uint4*)(Kp + 48 + lh * 8);
            v8f z = __builtin_amdgcn_wmma_f32_16x16x32_bf16(
                false, ka0.v, false, bq[0].v, (short)0, vz, false, false);
            st[sub] = __builtin_amdgcn_wmma_f32_16x16x32_bf16(
                false, ka1.v, false, bq[1].v, (short)0, z, false, false);
        }
        // Scale, causal mask, per-lane running softmax
        float p0[8], p1[8];
        float cmax = -1e30f;
#pragma unroll
        for (int r = 0; r < 8; ++r) {
            int kt0 = ktb + lh * 8 + r;
            int kt1 = ktb + 16 + lh * 8 + r;
            float v0 = st[0][r] * 0.125f;  // 1/sqrt(64)
            float v1 = st[1][r] * 0.125f;
            if (kt0 > q) v0 = -1e30f;
            if (kt1 > q) v1 = -1e30f;
            p0[r] = v0; p1[r] = v1;
            cmax = fmaxf(cmax, fmaxf(v0, v1));
        }
        cmax = fmaxf(cmax, __shfl_xor(cmax, 16, 32));  // merge lane halves
        float mnew = fmaxf(mrun, cmax);
        float scale = __expf(mrun - mnew);
        float psum = 0.f;
#pragma unroll
        for (int r = 0; r < 8; ++r) {
            p0[r] = __expf(p0[r] - mnew);
            p1[r] = __expf(p1[r] - mnew);
            psum += p0[r] + p1[r];
        }
        psum += __shfl_xor(psum, 16, 32);
        srun = srun * scale + psum;
        mrun = mnew;
#pragma unroll
        for (int dt = 0; dt < 4; ++dt)
#pragma unroll
            for (int r = 0; r < 8; ++r) acc[dt][r] *= scale;

        // Pack P^T into B-fragment (pure in-lane: C rows map onto B K-slots)
        BFrag pb;
#pragma unroll
        for (int e = 0; e < 8; ++e) pb.v[e] = (__bf16)p0[e];
#pragma unroll
        for (int e = 0; e < 8; ++e) pb.v[8 + e] = (__bf16)p1[e];

        // O^T += V^T_tile @ P^T
#pragma unroll
        for (int dt = 0; dt < 4; ++dt) {
            const __bf16* Vp = Vt + ((size_t)bh * Dcfg + dt * 16 + lr) * Tcfg + ktb;
            BFrag va;
            va.u[0] = *(const uint4*)(Vp + lh * 8);
            va.u[1] = *(const uint4*)(Vp + 16 + lh * 8);
            acc[dt] = __builtin_amdgcn_wmma_f32_16x16x32_bf16(
                false, va.v, false, pb.v, (short)0, acc[dt], false, false);
        }
    }

    // Normalize + store bf16 at [b, t=q, h*64 + d] — 8 consecutive d per lane
    float inv = 1.f / srun;
    int b = bh >> 4, h = bh & 15;
    __bf16* op = Outp + ((size_t)(b * Tcfg + q)) * Ecfg + h * Dcfg + 8 * lh;
#pragma unroll
    for (int dt = 0; dt < 4; ++dt) {
        unsigned pk[4];
#pragma unroll
        for (int i = 0; i < 4; ++i) {
            union { __bf16 h2[2]; unsigned u; } cv;
            cv.h2[0] = (__bf16)(acc[dt][2 * i] * inv);
            cv.h2[1] = (__bf16)(acc[dt][2 * i + 1] * inv);
            pk[i] = cv.u;
        }
        uint4 w = {pk[0], pk[1], pk[2], pk[3]};
        *(uint4*)(op + dt * 16) = w;
    }
}

// ---------------------------------------------------------------------------
// Launch
// ---------------------------------------------------------------------------
extern "C" void kernel_launch(void* const* d_in, const int* in_sizes, int n_in,
                              void* d_out, int out_size, void* d_ws, size_t ws_size,
                              hipStream_t stream) {
    const float* x      = (const float*)d_in[0];
    const float* ln1_g  = (const float*)d_in[1];
    const float* ln1_b  = (const float*)d_in[2];
    const float* w_qkv  = (const float*)d_in[3];
    const float* b_qkv  = (const float*)d_in[4];
    const float* w_out  = (const float*)d_in[5];
    const float* b_out  = (const float*)d_in[6];
    const float* ln2_g  = (const float*)d_in[7];
    const float* ln2_b  = (const float*)d_in[8];
    const float* w_fc   = (const float*)d_in[9];
    const float* b_fc   = (const float*)d_in[10];
    const float* w_proj = (const float*)d_in[11];
    const float* b_proj = (const float*)d_in[12];

    // Workspace carve-up
    char* ws = (char*)d_ws;
    size_t o = 0;
    auto alloc = [&](size_t bytes) {
        char* p = ws + o;
        o = (o + bytes + 255) & ~(size_t)255;
        return p;
    };
    const size_t M = Mcfg, E = Ecfg;
    __bf16* wqkvT  = (__bf16*)alloc(3 * E * E * 2);        // [3E, E]
    __bf16* woutT  = (__bf16*)alloc(E * E * 2);            // [E, E]
    __bf16* wfcT   = (__bf16*)alloc(4 * E * E * 2);        // [4E, E]
    __bf16* wprojT = (__bf16*)alloc(E * 4 * E * 2);        // [E, 4E]
    __bf16* h1     = (__bf16*)alloc(M * E * 2);
    __bf16* qkv    = (__bf16*)alloc(M * 3 * E * 2);
    __bf16* Qb     = (__bf16*)alloc(M * E * 2);
    __bf16* Kb     = (__bf16*)alloc(M * E * 2);
    __bf16* Vtb    = (__bf16*)alloc(M * E * 2);
    __bf16* attn   = (__bf16*)alloc(M * E * 2);
    float*  x1     = (float*)alloc(M * E * 4);
    __bf16* h2     = (__bf16*)alloc(M * E * 2);
    __bf16* fc     = (__bf16*)alloc(M * 4 * E * 2);

    // 1) Weights -> bf16 transposed
    cvt_wT_kernel<<<(int)((E * 3 * E + 255) / 256), 256, 0, stream>>>(w_qkv, wqkvT, E, 3 * E);
    cvt_wT_kernel<<<(int)((E * E + 255) / 256),     256, 0, stream>>>(w_out, woutT, E, E);
    cvt_wT_kernel<<<(int)((E * 4 * E + 255) / 256), 256, 0, stream>>>(w_fc,  wfcT,  E, 4 * E);
    cvt_wT_kernel<<<(int)((4 * E * E + 255) / 256), 256, 0, stream>>>(w_proj, wprojT, 4 * E, E);

    // 2) LN1
    ln_kernel<<<(int)M, 256, 0, stream>>>(x, ln1_g, ln1_b, h1);

    // 3) QKV GEMM: [M,E] x [E,3E] -> bf16 [M,3E]
    gemm_kernel<0><<<dim3(3 * E / 128, M / 128), 256, 0, stream>>>(
        h1, wqkvT, b_qkv, nullptr, qkv, M, 3 * E, E);

    // 4) Split heads (V transposed)
    split_qkv_kernel<<<(int)(M * E / 256), 256, 0, stream>>>(qkv, Qb, Kb, Vtb);

    // 5) Flash attention: B*H*(T/16) = 4096 waves = 512 blocks
    attn_kernel<<<Bcfg * Hcfg * (Tcfg / 16) / 8, 256, 0, stream>>>(Qb, Kb, Vtb, attn);

    // 6) Out-proj + residual(x) -> x1 (f32)
    gemm_kernel<1><<<dim3(E / 128, M / 128), 256, 0, stream>>>(
        attn, woutT, b_out, x, x1, M, E, E);

    // 7) LN2
    ln_kernel<<<(int)M, 256, 0, stream>>>(x1, ln2_g, ln2_b, h2);

    // 8) FC + GELU -> bf16 [M,4E]
    gemm_kernel<2><<<dim3(4 * E / 128, M / 128), 256, 0, stream>>>(
        h2, wfcT, b_fc, nullptr, fc, M, 4 * E, E);

    // 9) Proj + residual(x1) -> d_out (f32)
    gemm_kernel<1><<<dim3(E / 128, M / 128), 256, 0, stream>>>(
        fc, wprojT, b_proj, x1, (float*)d_out, M, E, 4 * E);
}
